// AttentionPairBias_85899345920319
// MI455X (gfx1250) — matrix operand order repaired and checked by reading, not compiled, not run on hardware
//
#include <hip/hip_runtime.h>
#include <cstdint>
#include <cstddef>

typedef __bf16 bf16;
typedef __attribute__((ext_vector_type(16))) __bf16 v16bf;
typedef __attribute__((ext_vector_type(8)))  float  v8f;

static constexpr int NT = 768;   // tokens
static constexpr int ED = 768;   // embed dim (== H*D)
static constexpr int SE = 384;   // s dim
static constexpr int PE = 128;   // pair dim
static constexpr int NH = 16;    // heads
static constexpr int HD = 48;    // head dim
static constexpr int DPAD = 64;  // head dim padded to WMMA K multiple

__device__ __forceinline__ float sigmoidf_(float x) {
  return 1.0f / (1.0f + __expf(-x));
}

__device__ __forceinline__ v8f wmma_bf16(v16bf a, v16bf b, v8f c) {
  return __builtin_amdgcn_wmma_f32_16x16x32_bf16(false, a, false, b, (short)0, c,
                                                 false, false);
}

// A-fragment (16x32 bf16, MxK), A row-major. Per ISA 7.12.2:
// lanes 0-15: M=lane, K chunks {0..7,16..23}; lanes 16-31: M=lane-16, {8..15,24..31}
__device__ __forceinline__ v16bf load_frag_a(const bf16* A, int lda, int row0,
                                             int k0, int lane) {
  int m = lane & 15;
  int kb = (lane & 16) ? 8 : 0;
  const bf16* p = A + (size_t)(row0 + m) * lda + k0 + kb;
  union { v16bf v; uint4 q[2]; } u;
  u.q[0] = *(const uint4*)(p);        // K = kb .. kb+7
  u.q[1] = *(const uint4*)(p + 16);   // K = kb+16 .. kb+23
  return u.v;
}

// B-fragment (32x16 bf16, KxN), B stored transposed (Bt: N rows x K cols, row-major).
// lanes 0-15: N=lane, K chunks {0..7,16..23}; lanes 16-31: N=lane-16, {8..15,24..31}.
__device__ __forceinline__ v16bf load_frag_b(const bf16* Bt, int ldb, int col0,
                                             int k0, int lane) {
  int n = lane & 15;
  int kb = (lane & 16) ? 8 : 0;
  const bf16* p = Bt + (size_t)(col0 + n) * ldb + k0 + kb;
  union { v16bf v; uint4 q[2]; } u;
  u.q[0] = *(const uint4*)(p);
  u.q[1] = *(const uint4*)(p + 16);
  return u.v;
}

// ---- register-blocked bf16 WMMA GEMM: each wave owns a (16*RT) x (16*CT) C tile
//      C[M,Nc] = A[M,K] * Bt[Nc,K]^T ; A frags reused CT times, B frags RT times.
template <int RT, int CT>
__global__ void k_gemm_blk(const bf16* __restrict__ A, int lda, long long sA,
                           const bf16* __restrict__ Bt, int ldb, long long sB,
                           float* __restrict__ C, int ldc, long long sC,
                           int M, int K, int tilesPerRow) {
  int wave = threadIdx.x >> 5;
  int lane = threadIdx.x & 31;
  int tile = blockIdx.x * (blockDim.x >> 5) + wave;
  int ntiles = (M / (16 * RT)) * tilesPerRow;
  if (tile >= ntiles) return;            // wave-uniform: EXEC stays all-1s
  int row0 = (tile / tilesPerRow) * (16 * RT);
  int col0 = (tile % tilesPerRow) * (16 * CT);
  const bf16* Ab = A + sA * blockIdx.z;
  const bf16* Bb = Bt + sB * blockIdx.z;
  float* Cb = C + sC * blockIdx.z;

  v8f acc[RT][CT];
#pragma unroll
  for (int r = 0; r < RT; ++r)
#pragma unroll
    for (int c = 0; c < CT; ++c)
      acc[r][c] = (v8f){0.f, 0.f, 0.f, 0.f, 0.f, 0.f, 0.f, 0.f};

  for (int k0 = 0; k0 < K; k0 += 32) {
    v16bf af[RT];
#pragma unroll
    for (int r = 0; r < RT; ++r)
      af[r] = load_frag_a(Ab, lda, row0 + 16 * r, k0, lane);
    v16bf bf_[CT];
#pragma unroll
    for (int c = 0; c < CT; ++c)
      bf_[c] = load_frag_b(Bb, ldb, col0 + 16 * c, k0, lane);
#pragma unroll
    for (int r = 0; r < RT; ++r)
#pragma unroll
      for (int c = 0; c < CT; ++c)
        acc[r][c] = wmma_bf16(af[r], bf_[c], acc[r][c]);
  }

  int hi = (lane & 16) ? 8 : 0;
#pragma unroll
  for (int r = 0; r < RT; ++r) {
#pragma unroll
    for (int c = 0; c < CT; ++c) {
      float* pc = Cb + (size_t)(row0 + 16 * r + hi) * ldc + col0 + 16 * c + (lane & 15);
#pragma unroll
      for (int rr = 0; rr < 8; ++rr) pc[(size_t)rr * ldc] = acc[r][c][rr];
    }
  }
}

// ---------------- weight convert + transpose: f32[r,c] -> bf16[c,r] ----------
__global__ void k_cvt_T(const float* __restrict__ src, bf16* __restrict__ dst,
                        int rows, int cols) {
  int i = blockIdx.x * blockDim.x + threadIdx.x;
  if (i >= rows * cols) return;
  int r = i / cols, c = i % cols;
  dst[(size_t)c * rows + r] = (bf16)src[i];
}

// ---------------- LayerNorm of s: writes s_n (bf16) and raw s (bf16) ---------
__global__ void k_ln_s(const float* __restrict__ s, bf16* __restrict__ s_n,
                       bf16* __restrict__ s_raw) {
  int n = blockIdx.x, t = threadIdx.x;            // 128 threads
  __shared__ float ssum[128], ssq[128];
  float ls = 0.f, lq = 0.f;
  for (int j = t; j < SE; j += 128) {
    float x = s[(size_t)n * SE + j];
    ls += x; lq += x * x;
  }
  ssum[t] = ls; ssq[t] = lq; __syncthreads();
  for (int off = 64; off > 0; off >>= 1) {
    if (t < off) { ssum[t] += ssum[t + off]; ssq[t] += ssq[t + off]; }
    __syncthreads();
  }
  float mu = ssum[0] * (1.0f / SE);
  float var = ssq[0] * (1.0f / SE) - mu * mu;
  float rs = rsqrtf(var + 1e-5f);
  for (int j = t; j < SE; j += 128) {
    float x = s[(size_t)n * SE + j];
    s_n[(size_t)n * SE + j] = (bf16)((x - mu) * rs);
    s_raw[(size_t)n * SE + j] = (bf16)x;
  }
}

// ---------------- LayerNorm of a (no affine): writes a_n (f32) ---------------
__global__ void k_ln_a(const float* __restrict__ a, float* __restrict__ a_n) {
  int n = blockIdx.x, t = threadIdx.x;            // 256 threads
  __shared__ float ssum[256], ssq[256];
  float ls = 0.f, lq = 0.f;
  for (int j = t; j < ED; j += 256) {
    float x = a[(size_t)n * ED + j];
    ls += x; lq += x * x;
  }
  ssum[t] = ls; ssq[t] = lq; __syncthreads();
  for (int off = 128; off > 0; off >>= 1) {
    if (t < off) { ssum[t] += ssum[t + off]; ssq[t] += ssq[t + off]; }
    __syncthreads();
  }
  float mu = ssum[0] * (1.0f / ED);
  float var = ssq[0] * (1.0f / ED) - mu * mu;
  float rs = rsqrtf(var + 1e-5f);
  for (int j = t; j < ED; j += 256)
    a_n[(size_t)n * ED + j] = (a[(size_t)n * ED + j] - mu) * rs;
}

// ---------------- a_mod = sigmoid(gl) * a_n + sh  (bf16) ---------------------
__global__ void k_amod(const float* __restrict__ gl, const float* __restrict__ sh,
                       const float* __restrict__ a_n, bf16* __restrict__ amod) {
  int i = blockIdx.x * blockDim.x + threadIdx.x;
  if (i >= NT * ED) return;
  amod[i] = (bf16)(sigmoidf_(gl[i]) * a_n[i] + sh[i]);
}

// ------- pack q(+bias,*scale)/k into [H][N][64] bf16 (zero pad), vT, gate ----
__global__ void k_qkvg_pack(const float* __restrict__ qf, const float* __restrict__ kf,
                            const float* __restrict__ vf, const float* __restrict__ gf,
                            const float* __restrict__ bq,
                            bf16* __restrict__ qb, bf16* __restrict__ kb,
                            bf16* __restrict__ vT, float* __restrict__ gate) {
  int i = blockIdx.x * blockDim.x + threadIdx.x;   // NT * 1024 threads
  int n = i >> 10, j = i & 1023;
  if (n >= NT) return;
  const float scale = 0.14433756729740643f;        // 1/sqrt(48)
  if (j < ED) {
    int h = j / HD, d = j % HD;
    size_t src = (size_t)n * ED + j;
    qb[((size_t)h * NT + n) * DPAD + d] = (bf16)((qf[src] + bq[j]) * scale);
    kb[((size_t)h * NT + n) * DPAD + d] = (bf16)kf[src];
    vT[((size_t)h * HD + d) * NT + n] = (bf16)vf[src];
    gate[src] = sigmoidf_(gf[src]);
  } else {                                         // zero the D-padding 48..63
    int p = j - ED;                                // 256 pad slots = 16 heads x 16
    int h = p >> 4, d = HD + (p & 15);
    qb[((size_t)h * NT + n) * DPAD + d] = (bf16)0.0f;
    kb[((size_t)h * NT + n) * DPAD + d] = (bf16)0.0f;
  }
}

// ------- fused pair bias: LN(pair[n,m,:]) @ w_pair^T + beta -> bf16[H][N][N] -
__global__ void k_pairbias(const float* __restrict__ pair, const float* __restrict__ beta,
                           const float* __restrict__ gvec, const float* __restrict__ bvec,
                           const bf16* __restrict__ wpT, bf16* __restrict__ biasb) {
  int wave = threadIdx.x >> 5, lane = threadIdx.x & 31;
  int idx = blockIdx.x * 8 + wave;                 // NT * 48 tiles total
  int n = idx / (NT >> 4);
  int m0 = (idx % (NT >> 4)) << 4;
  int mrow = lane & 15;
  const float* prow = pair + ((size_t)n * NT + m0 + mrow) * PE;

  // Each lane loads exactly the 64 channels its A-fragments need.
  float xs[64];
  float ls = 0.f, lq = 0.f;
#pragma unroll
  for (int t = 0; t < 4; ++t) {
    int base = 32 * t + ((lane & 16) ? 8 : 0);
#pragma unroll
    for (int c = 0; c < 8; ++c) {
      float x0 = prow[base + c];
      float x1 = prow[base + 16 + c];
      xs[t * 16 + c] = x0; xs[t * 16 + 8 + c] = x1;
      ls += x0 + x1; lq += x0 * x0 + x1 * x1;
    }
  }
  // partner lane (^16) holds the other half of this row
  ls += __shfl_xor(ls, 16, 32);
  lq += __shfl_xor(lq, 16, 32);
  float mu = ls * (1.0f / PE);
  float var = lq * (1.0f / PE) - mu * mu;
  float rs = rsqrtf(var + 1e-5f);

  v8f acc = {0.f, 0.f, 0.f, 0.f, 0.f, 0.f, 0.f, 0.f};
#pragma unroll
  for (int t = 0; t < 4; ++t) {
    int base = 32 * t + ((lane & 16) ? 8 : 0);
    union { v16bf v; bf16 e[16]; } ua;
#pragma unroll
    for (int c = 0; c < 8; ++c) {
      int k0 = base + c, k1 = base + 16 + c;
      ua.e[c]     = (bf16)((xs[t * 16 + c]     - mu) * rs * gvec[k0] + bvec[k0]);
      ua.e[8 + c] = (bf16)((xs[t * 16 + 8 + c] - mu) * rs * gvec[k1] + bvec[k1]);
    }
    v16bf bfr = load_frag_b(wpT, PE, 0, 32 * t, lane);
    acc = wmma_bf16(ua.v, bfr, acc);
  }
  // C: rows = m (8 consecutive per lane), cols = h
  int h = lane & 15;
  int hi = (lane & 16) ? 8 : 0;
  const float* bp = beta + (size_t)n * NT + m0 + hi;
  bf16* op = biasb + ((size_t)h * NT + n) * NT + m0 + hi;
#pragma unroll
  for (int r = 0; r < 8; ++r) op[r] = (bf16)(acc[r] + bp[r]);
}

// ------- logits (transposed tile so bias add is contiguous): in-place bf16 ---
__global__ void k_logits(const bf16* __restrict__ qb, const bf16* __restrict__ kb,
                         bf16* __restrict__ wbuf) {
  int h = blockIdx.z;
  int wave = threadIdx.x >> 5, lane = threadIdx.x & 31;
  int tile = blockIdx.x * 8 + wave;                // (NT/16)^2 tiles per head
  int mt = tile / (NT >> 4), ntl = tile % (NT >> 4);
  int m0 = mt << 4, n0 = ntl << 4;
  const bf16* A = kb + (size_t)h * NT * DPAD;      // rows = m (k tokens)
  const bf16* Bt = qb + (size_t)h * NT * DPAD;     // rows = n (q tokens)
  v8f acc = {0.f, 0.f, 0.f, 0.f, 0.f, 0.f, 0.f, 0.f};
#pragma unroll
  for (int t = 0; t < 2; ++t) {
    v16bf a = load_frag_a(A, DPAD, m0, 32 * t, lane);
    v16bf b = load_frag_b(Bt, DPAD, n0, 32 * t, lane);
    acc = wmma_bf16(a, b, acc);
  }
  int nn = n0 + (lane & 15);
  int hi = (lane & 16) ? 8 : 0;
  bf16* p = wbuf + ((size_t)h * NT + nn) * NT + m0 + hi;
#pragma unroll
  for (int r = 0; r < 8; ++r) p[r] = (bf16)(acc[r] + (float)p[r]);
}

// ------- row softmax over m, in place (bf16 logits -> bf16 weights) ----------
__global__ void k_softmax(bf16* __restrict__ wbuf) {
  int row = blockIdx.x, t = threadIdx.x;           // 256 threads, NT cols
  bf16* p = wbuf + (size_t)row * NT;
  __shared__ float sred[256];
  float lmax = -3.0e38f;
  for (int j = t; j < NT; j += 256) lmax = fmaxf(lmax, (float)p[j]);
  sred[t] = lmax; __syncthreads();
  for (int off = 128; off > 0; off >>= 1) {
    if (t < off) sred[t] = fmaxf(sred[t], sred[t + off]);
    __syncthreads();
  }
  float mx = sred[0]; __syncthreads();
  float vals[3]; float lsum = 0.f; int c = 0;
  for (int j = t; j < NT; j += 256, ++c) {
    float e = __expf((float)p[j] - mx);
    vals[c] = e; lsum += e;
  }
  sred[t] = lsum; __syncthreads();
  for (int off = 128; off > 0; off >>= 1) {
    if (t < off) sred[t] += sred[t + off];
    __syncthreads();
  }
  float inv = 1.0f / sred[0];
  c = 0;
  for (int j = t; j < NT; j += 256, ++c) p[j] = (bf16)(vals[c] * inv);
}

// ------- gate: ga = attn * gate (bf16) ---------------------------------------
__global__ void k_gatemul(const float* __restrict__ attn, const float* __restrict__ gate,
                          bf16* __restrict__ ga) {
  int i = blockIdx.x * blockDim.x + threadIdx.x;
  if (i >= NT * ED) return;
  ga[i] = (bf16)(attn[i] * gate[i]);
}

// ------- final: out = sigmoid(sg + b_out) * out1 -----------------------------
__global__ void k_final(const float* __restrict__ out1, const float* __restrict__ sg,
                        const float* __restrict__ b_out, float* __restrict__ out) {
  int i = blockIdx.x * blockDim.x + threadIdx.x;
  if (i >= NT * ED) return;
  out[i] = sigmoidf_(sg[i] + b_out[i % ED]) * out1[i];
}

// =============================================================================
extern "C" void kernel_launch(void* const* d_in, const int* in_sizes, int n_in,
                              void* d_out, int out_size, void* d_ws, size_t ws_size,
                              hipStream_t stream) {
  (void)in_sizes; (void)n_in; (void)out_size; (void)ws_size;
  const float* a        = (const float*)d_in[0];
  const float* s        = (const float*)d_in[1];
  const float* pair     = (const float*)d_in[2];
  const float* beta     = (const float*)d_in[3];
  const float* wadag    = (const float*)d_in[4];
  const float* wadab    = (const float*)d_in[5];
  const float* wq       = (const float*)d_in[6];
  const float* bq       = (const float*)d_in[7];
  const float* wk       = (const float*)d_in[8];
  const float* wv       = (const float*)d_in[9];
  const float* ln_pg    = (const float*)d_in[10];
  const float* ln_pb    = (const float*)d_in[11];
  const float* w_pair   = (const float*)d_in[12];
  const float* w_gate   = (const float*)d_in[13];
  const float* w_proj   = (const float*)d_in[14];
  const float* w_out    = (const float*)d_in[15];
  const float* b_out    = (const float*)d_in[16];
  float* out = (float*)d_out;

  // ---- workspace bump allocator (256B aligned) ----
  char* w = (char*)d_ws;
  auto alloc = [&](size_t bytes) -> char* {
    char* p = w; w += (bytes + 255) & ~(size_t)255; return p;
  };
  bf16* s_n    = (bf16*)alloc((size_t)NT * SE * 2);
  bf16* s_bf   = (bf16*)alloc((size_t)NT * SE * 2);
  float* a_n   = (float*)alloc((size_t)NT * ED * 4);
  bf16* wadagT = (bf16*)alloc((size_t)ED * SE * 2);
  bf16* wadabT = (bf16*)alloc((size_t)ED * SE * 2);
  bf16* wqT    = (bf16*)alloc((size_t)ED * ED * 2);
  bf16* wkT    = (bf16*)alloc((size_t)ED * ED * 2);
  bf16* wvT    = (bf16*)alloc((size_t)ED * ED * 2);
  bf16* wgT    = (bf16*)alloc((size_t)ED * ED * 2);
  bf16* wprojT = (bf16*)alloc((size_t)ED * ED * 2);
  bf16* woutT  = (bf16*)alloc((size_t)ED * SE * 2);
  bf16* wpT    = (bf16*)alloc((size_t)NH * PE * 2);
  float* glf   = (float*)alloc((size_t)NT * ED * 4);
  float* shf   = (float*)alloc((size_t)NT * ED * 4);
  bf16* amod   = (bf16*)alloc((size_t)NT * ED * 2);
  float* qf    = (float*)alloc((size_t)NT * ED * 4);
  float* kf    = (float*)alloc((size_t)NT * ED * 4);
  float* vf    = (float*)alloc((size_t)NT * ED * 4);
  float* gf    = (float*)alloc((size_t)NT * ED * 4);
  bf16* qb     = (bf16*)alloc((size_t)NH * NT * DPAD * 2);
  bf16* kb     = (bf16*)alloc((size_t)NH * NT * DPAD * 2);
  bf16* vT     = (bf16*)alloc((size_t)NH * HD * NT * 2);
  float* gate  = (float*)alloc((size_t)NT * ED * 4);
  bf16* wbuf   = (bf16*)alloc((size_t)NH * NT * NT * 2);   // bias -> logits -> weights
  float* attn  = (float*)alloc((size_t)NT * ED * 4);
  bf16* ga     = (bf16*)alloc((size_t)NT * ED * 2);
  float* out1  = (float*)alloc((size_t)NT * ED * 4);
  float* sg    = (float*)alloc((size_t)NT * ED * 4);

  auto cvtT = [&](const float* src, bf16* dst, int rows, int cols) {
    int n = rows * cols;
    k_cvt_T<<<(n + 255) / 256, 256, 0, stream>>>(src, dst, rows, cols);
  };
  // 32x64 wave tile (RT=2, CT=4): Nc must be a multiple of 64
  auto gemm64 = [&](const bf16* A, int lda, long long sA, const bf16* Bt, int ldb,
                    long long sB, float* C, int ldc, long long sC, int M, int Nc,
                    int K, int batch) {
    int tpr = Nc / 64;
    int tiles = (M / 32) * tpr;
    dim3 g((tiles + 7) / 8, 1, batch);
    k_gemm_blk<2, 4><<<g, 256, 0, stream>>>(A, lda, sA, Bt, ldb, sB, C, ldc, sC,
                                            M, K, tpr);
  };
  // 32x48 wave tile (RT=2, CT=3): for the per-head PV GEMM (Nc = 48)
  auto gemm48 = [&](const bf16* A, int lda, long long sA, const bf16* Bt, int ldb,
                    long long sB, float* C, int ldc, long long sC, int M, int Nc,
                    int K, int batch) {
    int tpr = Nc / 48;
    int tiles = (M / 32) * tpr;
    dim3 g((tiles + 7) / 8, 1, batch);
    k_gemm_blk<2, 3><<<g, 256, 0, stream>>>(A, lda, sA, Bt, ldb, sB, C, ldc, sC,
                                            M, K, tpr);
  };

  // 1) weights -> bf16 transposed
  cvtT(wadag, wadagT, SE, ED);
  cvtT(wadab, wadabT, SE, ED);
  cvtT(wq, wqT, ED, ED);
  cvtT(wk, wkT, ED, ED);
  cvtT(wv, wvT, ED, ED);
  cvtT(w_gate, wgT, ED, ED);
  cvtT(w_proj, wprojT, ED, ED);
  cvtT(w_out, woutT, SE, ED);
  cvtT(w_pair, wpT, PE, NH);

  // 2) layernorms
  k_ln_s<<<NT, 128, 0, stream>>>(s, s_n, s_bf);
  k_ln_a<<<NT, 256, 0, stream>>>(a, a_n);

  // 3) AdaLN conditioning GEMMs + a_mod
  gemm64(s_n, SE, 0, wadagT, SE, 0, glf, ED, 0, NT, ED, SE, 1);
  gemm64(s_n, SE, 0, wadabT, SE, 0, shf, ED, 0, NT, ED, SE, 1);
  k_amod<<<(NT * ED + 255) / 256, 256, 0, stream>>>(glf, shf, a_n, amod);

  // 4) q/k/v/gate projections (E x E each) + pack
  gemm64(amod, ED, 0, wqT, ED, 0, qf, ED, 0, NT, ED, ED, 1);
  gemm64(amod, ED, 0, wkT, ED, 0, kf, ED, 0, NT, ED, ED, 1);
  gemm64(amod, ED, 0, wvT, ED, 0, vf, ED, 0, NT, ED, ED, 1);
  gemm64(amod, ED, 0, wgT, ED, 0, gf, ED, 0, NT, ED, ED, 1);
  k_qkvg_pack<<<(NT * 1024 + 255) / 256, 256, 0, stream>>>(qf, kf, vf, gf, bq,
                                                           qb, kb, vT, gate);

  // 5) fused pair-bias (streams the 302MB pair tensor exactly once)
  k_pairbias<<<(NT * (NT / 16)) / 8, 256, 0, stream>>>(pair, beta, ln_pg, ln_pb,
                                                       wpT, wbuf);

  // 6) logits (adds bias in place), softmax in place
  {
    int tiles = (NT >> 4) * (NT >> 4);
    dim3 g(tiles / 8, 1, NH);
    k_logits<<<g, 256, 0, stream>>>(qb, kb, wbuf);
  }
  k_softmax<<<NH * NT, 256, 0, stream>>>(wbuf);

  // 7) attn = weights @ v  (batched over heads; C strided into [N, H*D])
  gemm48(wbuf, NT, (long long)NT * NT, vT, NT, (long long)HD * NT,
         attn, ED, (long long)HD, NT, HD, NT, NH);

  // 8) gate, output projection, s-gate, final
  k_gatemul<<<(NT * ED + 255) / 256, 256, 0, stream>>>(attn, gate, ga);
  gemm64(ga, ED, 0, wprojT, ED, 0, out1, ED, 0, NT, ED, ED, 1);
  gemm64(s_bf, SE, 0, woutT, SE, 0, sg, ED, 0, NT, ED, SE, 1);
  k_final<<<(NT * ED + 255) / 256, 256, 0, stream>>>(out1, sg, b_out, out);
}
